// ResNet_89103391523579
// MI455X (gfx1250) — 8-bit matrix operand order repaired and checked by reading, compiled, not run on hardware
//
#include <hip/hip_runtime.h>
#include <stdint.h>

typedef int v8i __attribute__((ext_vector_type(8)));

#if defined(__has_builtin)
#if __has_builtin(__builtin_amdgcn_sched_barrier)
#define SCHED_BARRIER() __builtin_amdgcn_sched_barrier(0)
#endif
#endif
#ifndef SCHED_BARRIER
#define SCHED_BARRIER()
#endif

#define NI   16
#define C    64
#define H    56
#define W    56
#define PIMG (H*W)
#define PIX  (NI*PIMG)
#define PH   58
#define PW   58
#define PADB (NI*PH*PW*C)
#define NOUT (NI*C*H*W)
#define WELEM (C*C*9)
#define WPKB  36864

#define O_MX    0
#define O_MW1   1
#define O_MW2   2
#define O_MR1   3
#define O_MOUT  4
#define O_HMACT 5
#define O_SX    16
#define O_SQ1   (O_SX  + 9*512)
#define O_WB1   (O_SQ1 + 9*512)
#define O_WB2   (O_WB1 + 4608)
#define CNT_U32 (O_WB2 + 4608)
#define CNT_BYTES 73984

__device__ __forceinline__ int chpos(int c) {
    return ((c & 8) << 2) | ((c & 48) >> 1) | (c & 7);
}

__global__ void k_zero(uint32_t* __restrict__ p, int n32) {
    int i = blockIdx.x * blockDim.x + threadIdx.x;
    if (i < n32) p[i] = 0u;
}

__global__ void k_absmax(const float* __restrict__ x, int n, uint32_t* __restrict__ slot) {
    float m = 0.f;
    for (int i = blockIdx.x * blockDim.x + threadIdx.x; i < n; i += gridDim.x * blockDim.x)
        m = fmaxf(m, fabsf(x[i]));
    atomicMax(slot, __float_as_uint(m));
}

__global__ void k_quant_w(const float* __restrict__ w, const uint32_t* __restrict__ cnt,
                          int mslot, int8_t* __restrict__ packed, uint32_t* __restrict__ wbits) {
    int i = blockIdx.x * blockDim.x + threadIdx.x;
    if (i >= WELEM) return;
    float mw = __uint_as_float(cnt[mslot]);
    float scale = mw * (1.f / 127.f);
    float inv = scale > 0.f ? 1.f / scale : 0.f;
    int o = i / (C * 9);
    int rem = i % (C * 9);
    int c = rem / 9;
    int tap = rem % 9;
    float qf = rintf(w[i] * inv);
    qf = fminf(fmaxf(qf, -127.f), 127.f);
    int q = (int)qf;
    int g = o >> 4, ng = o & 15;
    int lane = ng + (((c >> 3) & 1) ? 16 : 0);
    int byt  = (c & 7) + ((c >> 4) << 3);
    packed[((tap * 4 + g) * 32 + lane) * 32 + byt] = (int8_t)q;
    unsigned u = (unsigned)q & 0xFFu;
    #pragma unroll
    for (int k = 0; k < 8; k++)
        if ((u >> k) & 1u) atomicAdd(&wbits[(c * 8 + k) * 9 + tap], 1u);
}

__global__ void k_quant_x(const float* __restrict__ x, uint32_t* __restrict__ cnt,
                          int8_t* __restrict__ qpad) {
    __shared__ uint32_t sT[512];
    __shared__ uint32_t sAct;
    for (int j = threadIdx.x; j < 512; j += blockDim.x) sT[j] = 0u;
    if (threadIdx.x == 0) sAct = 0u;
    __syncthreads();
    int i = blockIdx.x * blockDim.x + threadIdx.x;
    if (i < NOUT) {
        float scale = __uint_as_float(cnt[O_MX]) * (1.f / 127.f);
        float inv = scale > 0.f ? 1.f / scale : 0.f;
        int n = i / (C * PIMG);
        int r = i % (C * PIMG);
        int c = r / PIMG;
        int s = r % PIMG;
        int h = s / W, ww = s % W;
        float qf = rintf(x[i] * inv);
        qf = fminf(fmaxf(qf, -127.f), 127.f);
        int q = (int)qf;
        qpad[((((size_t)n * PH + h + 1) * PW + ww + 1) << 6) + chpos(c)] = (int8_t)q;
        unsigned u = (unsigned)q & 0xFFu;
        atomicAdd(&sAct, (unsigned)__popc(u));
        int e = c * 8;
        #pragma unroll
        for (int k = 0; k < 8; k++) {
            if ((u >> k) & 1u) {
                atomicAdd(&sT[e + k], 1u);
                if (h == 0)     atomicAdd(&cnt[O_SX + 1 * 512 + e + k], 1u);
                if (h == H - 1) atomicAdd(&cnt[O_SX + 2 * 512 + e + k], 1u);
                if (ww == 0)    atomicAdd(&cnt[O_SX + 3 * 512 + e + k], 1u);
                if (ww == W - 1)atomicAdd(&cnt[O_SX + 4 * 512 + e + k], 1u);
                if (h == 0     && ww == 0)     atomicAdd(&cnt[O_SX + 5 * 512 + e + k], 1u);
                if (h == 0     && ww == W - 1) atomicAdd(&cnt[O_SX + 6 * 512 + e + k], 1u);
                if (h == H - 1 && ww == 0)     atomicAdd(&cnt[O_SX + 7 * 512 + e + k], 1u);
                if (h == H - 1 && ww == W - 1) atomicAdd(&cnt[O_SX + 8 * 512 + e + k], 1u);
            }
        }
    }
    __syncthreads();
    for (int j = threadIdx.x; j < 512; j += blockDim.x)
        if (sT[j]) atomicAdd(&cnt[O_SX + j], sT[j]);
    if (threadIdx.x == 0 && sAct) atomicAdd(&cnt[O_HMACT], sAct);
}

__global__ void k_quant_q1(const float* __restrict__ obuf, uint32_t* __restrict__ cnt,
                           int8_t* __restrict__ qpad) {
    __shared__ uint32_t sT[512];
    __shared__ uint32_t sAct;
    for (int j = threadIdx.x; j < 512; j += blockDim.x) sT[j] = 0u;
    if (threadIdx.x == 0) sAct = 0u;
    __syncthreads();
    int i = blockIdx.x * blockDim.x + threadIdx.x;
    if (i < PIX * C) {
        float scale = __uint_as_float(cnt[O_MR1]) * (1.f / 255.f);
        float inv = scale > 0.f ? 1.f / scale : 0.f;
        int p = i >> 6, c = i & 63;
        int n = p / PIMG, s = p % PIMG, h = s / W, ww = s % W;
        float r = fmaxf(obuf[i], 0.f);
        float qf = fminf(rintf(r * inv), 255.f);
        int q = (int)qf;
        qpad[((((size_t)n * PH + h + 1) * PW + ww + 1) << 6) + chpos(c)] = (int8_t)q;
        unsigned u = (unsigned)q & 0xFFu;
        atomicAdd(&sAct, (unsigned)__popc(u));
        int e = c * 8;
        #pragma unroll
        for (int k = 0; k < 8; k++) {
            if ((u >> k) & 1u) {
                atomicAdd(&sT[e + k], 1u);
                if (h == 0)     atomicAdd(&cnt[O_SQ1 + 1 * 512 + e + k], 1u);
                if (h == H - 1) atomicAdd(&cnt[O_SQ1 + 2 * 512 + e + k], 1u);
                if (ww == 0)    atomicAdd(&cnt[O_SQ1 + 3 * 512 + e + k], 1u);
                if (ww == W - 1)atomicAdd(&cnt[O_SQ1 + 4 * 512 + e + k], 1u);
                if (h == 0     && ww == 0)     atomicAdd(&cnt[O_SQ1 + 5 * 512 + e + k], 1u);
                if (h == 0     && ww == W - 1) atomicAdd(&cnt[O_SQ1 + 6 * 512 + e + k], 1u);
                if (h == H - 1 && ww == 0)     atomicAdd(&cnt[O_SQ1 + 7 * 512 + e + k], 1u);
                if (h == H - 1 && ww == W - 1) atomicAdd(&cnt[O_SQ1 + 8 * 512 + e + k], 1u);
            }
        }
    }
    __syncthreads();
    for (int j = threadIdx.x; j < 512; j += blockDim.x)
        if (sT[j]) atomicAdd(&cnt[O_SQ1 + j], sT[j]);
    if (threadIdx.x == 0 && sAct) atomicAdd(&cnt[O_HMACT], sAct);
}

__device__ __forceinline__ v8i load_A(const int8_t* ap) {
    int4 a0 = *(const int4*)(ap);
    int4 a1 = *(const int4*)(ap + 16);
    return v8i{a0.x, a0.y, a0.z, a0.w, a1.x, a1.y, a1.z, a1.w};
}
__device__ __forceinline__ v8i load_B(const int8_t* wl, int t, int gi, int lane) {
    const int4* bp = (const int4*)(wl + ((t * 4 + gi) * 32 + lane) * 32);
    int4 b0 = bp[0];
    int4 b1 = bp[1];
    return v8i{b0.x, b0.y, b0.z, b0.w, b1.x, b1.y, b1.z, b1.w};
}

template <int MODE>
__global__ void __launch_bounds__(256)
k_conv(const int8_t* __restrict__ apad, const int8_t* __restrict__ wpk,
       const uint32_t* __restrict__ cnt, uint32_t* __restrict__ mslot,
       const float* __restrict__ gg, const float* __restrict__ bb,
       const float* __restrict__ mm, const float* __restrict__ vv,
       const int8_t* __restrict__ idpad, float* __restrict__ outbuf) {
    __shared__ int8_t wl[WPKB];
    {
        const int4* src = (const int4*)wpk;
        int4* dst = (int4*)wl;
        for (int j = threadIdx.x; j < WPKB / 16; j += blockDim.x) dst[j] = src[j];
    }
    __syncthreads();

    int lane = threadIdx.x & 31;
    int wave = threadIdx.x >> 5;
    int grp = blockIdx.x * 8 + wave;
    int pg = grp << 4;
    int lp = pg + (lane & 15);
    int n = lp / PIMG, s = lp % PIMG, h = s / W, ww = s % W;

    const int8_t* abase = apad + ((((size_t)n * PH + h) * PW + ww) << 6) + ((lane >> 4) << 5);

    v8i acc[4] = {v8i{0}, v8i{0}, v8i{0}, v8i{0}};

    v8i Ac = load_A(abase);
    v8i Bc0 = load_B(wl, 0, 0, lane);
    v8i Bc1 = load_B(wl, 0, 1, lane);
    v8i Bc2 = load_B(wl, 0, 2, lane);
    v8i Bc3 = load_B(wl, 0, 3, lane);

    #pragma unroll
    for (int t = 0; t < 9; t++) {
        v8i An, Bn0, Bn1, Bn2, Bn3;
        if (t < 8) {
            int tn = t + 1;
            An  = load_A(abase + (((tn / 3) * PW + (tn % 3)) << 6));
            Bn0 = load_B(wl, tn, 0, lane);
            Bn1 = load_B(wl, tn, 1, lane);
            Bn2 = load_B(wl, tn, 2, lane);
            Bn3 = load_B(wl, tn, 3, lane);
        }
        SCHED_BARRIER();
        acc[0] = __builtin_amdgcn_wmma_i32_16x16x64_iu8((MODE == 0), Ac, true, Bc0, acc[0], false, false);
        acc[1] = __builtin_amdgcn_wmma_i32_16x16x64_iu8((MODE == 0), Ac, true, Bc1, acc[1], false, false);
        acc[2] = __builtin_amdgcn_wmma_i32_16x16x64_iu8((MODE == 0), Ac, true, Bc2, acc[2], false, false);
        acc[3] = __builtin_amdgcn_wmma_i32_16x16x64_iu8((MODE == 0), Ac, true, Bc3, acc[3], false, false);
        SCHED_BARRIER();
        if (t < 8) {
            Ac = An; Bc0 = Bn0; Bc1 = Bn1; Bc2 = Bn2; Bc3 = Bn3;
        }
    }

    float sA, scl_id = 0.f;
    if constexpr (MODE == 0) {
        sA = (__uint_as_float(cnt[O_MX]) * (1.f / 127.f)) *
             (__uint_as_float(cnt[O_MW1]) * (1.f / 127.f));
    } else {
        sA = (__uint_as_float(cnt[O_MR1]) * (1.f / 255.f)) *
             (__uint_as_float(cnt[O_MW2]) * (1.f / 127.f));
        scl_id = __uint_as_float(cnt[O_MX]) * (1.f / 127.f);
    }
    float lmax = 0.f;
    #pragma unroll
    for (int gi = 0; gi < 4; gi++) {
        int c = (gi << 4) + (lane & 15);
        float inv = gg[c] * rsqrtf(vv[c] + 1e-5f);
        float beta = bb[c] - mm[c] * inv;
        #pragma unroll
        for (int r = 0; r < 8; r++) {
            int M = (lane < 16) ? r : (r + 8);
            int pm = pg + M;
            float val = (float)acc[gi][r] * sA * inv + beta;
            if constexpr (MODE == 1) {
                int n2 = pm / PIMG, s2 = pm % PIMG, h2 = s2 / W, w2 = s2 % W;
                int8_t qv = idpad[((((size_t)n2 * PH + h2 + 1) * PW + w2 + 1) << 6) + chpos(c)];
                val += (float)qv * scl_id;
            }
            outbuf[((size_t)pm << 6) + c] = val;
            lmax = fmaxf(lmax, val);
        }
    }
    atomicMax(mslot, __float_as_uint(fmaxf(lmax, 0.f)));
}

__global__ void k_final(const float* __restrict__ tbuf, const uint32_t* __restrict__ cnt,
                        float* __restrict__ out) {
    int i = blockIdx.x * blockDim.x + threadIdx.x;
    if (i >= NOUT) return;
    float scale = __uint_as_float(cnt[O_MOUT]) * (1.f / 255.f);
    float inv = scale > 0.f ? 1.f / scale : 0.f;
    int n = i / (C * PIMG);
    int r = i % (C * PIMG);
    int c = r / PIMG;
    int s = r % PIMG;
    int p = n * PIMG + s;
    float t = fmaxf(tbuf[((size_t)p << 6) + c], 0.f);
    float q = fminf(rintf(t * inv), 255.f);
    out[i] = q * scale;
}

__global__ void k_hm(const uint32_t* __restrict__ cnt, float* __restrict__ out) {
    __shared__ double sd[512];
    int e = threadIdx.x;
    double acc = 0.0;
    #pragma unroll
    for (int pass = 0; pass < 2; pass++) {
        const uint32_t* st = cnt + (pass ? O_SQ1 : O_SX);
        const uint32_t* wb = cnt + (pass ? O_WB2 : O_WB1);
        uint32_t T = st[e], R0 = st[512 + e], R55 = st[1024 + e];
        uint32_t C0 = st[1536 + e], C55 = st[2048 + e];
        uint32_t X00 = st[2560 + e], X0W = st[3072 + e];
        uint32_t XH0 = st[3584 + e], XHW = st[4096 + e];
        #pragma unroll
        for (int tap = 0; tap < 9; tap++) {
            int ey = tap / 3 - 1, ex = tap % 3 - 1;
            uint32_t S = T;
            if (ey == 1)  S -= R0;
            if (ey == -1) S -= R55;
            if (ex == 1)  S -= C0;
            if (ex == -1) S -= C55;
            if (ey == 1 && ex == 1)   S += X00;
            if (ey == 1 && ex == -1)  S += X0W;
            if (ey == -1 && ex == 1)  S += XH0;
            if (ey == -1 && ex == -1) S += XHW;
            acc += (double)wb[e * 9 + tap] * (double)S;
        }
    }
    sd[e] = acc;
    __syncthreads();
    for (int off = 256; off > 0; off >>= 1) {
        if (e < off) sd[e] += sd[e + off];
        __syncthreads();
    }
    if (e == 0) {
        out[NOUT + 0] = (float)cnt[O_HMACT];
        out[NOUT + 1] = (float)sd[0];
    }
}

extern "C" void kernel_launch(void* const* d_in, const int* in_sizes, int n_in,
                              void* d_out, int out_size, void* d_ws, size_t ws_size,
                              hipStream_t stream) {
    const float* x  = (const float*)d_in[0];
    const float* w1 = (const float*)d_in[1];
    const float* w2 = (const float*)d_in[2];
    const float* g1 = (const float*)d_in[3];
    const float* b1 = (const float*)d_in[4];
    const float* m1 = (const float*)d_in[5];
    const float* v1 = (const float*)d_in[6];
    const float* g2 = (const float*)d_in[7];
    const float* b2 = (const float*)d_in[8];
    const float* m2 = (const float*)d_in[9];
    const float* v2 = (const float*)d_in[10];
    float* out = (float*)d_out;

    uint8_t* ws = (uint8_t*)d_ws;
    uint32_t* cnt   = (uint32_t*)ws;
    int8_t*  qx_pad = (int8_t*)(ws + CNT_BYTES);
    int8_t*  q1_pad = qx_pad + PADB;
    int8_t*  wpk1   = q1_pad + PADB;
    int8_t*  wpk2   = wpk1 + WPKB;
    float*   obuf   = (float*)(ws + CNT_BYTES + 2 * (size_t)PADB + 2 * WPKB);
    float*   tbuf   = obuf;

    k_zero<<<(CNT_U32 + 255) / 256, 256, 0, stream>>>(cnt, CNT_U32);
    k_zero<<<(PADB / 4 + 255) / 256, 256, 0, stream>>>((uint32_t*)qx_pad, PADB / 4);
    k_zero<<<(PADB / 4 + 255) / 256, 256, 0, stream>>>((uint32_t*)q1_pad, PADB / 4);

    k_absmax<<<256, 256, 0, stream>>>(x, NOUT, cnt + O_MX);
    k_absmax<<<32, 256, 0, stream>>>(w1, WELEM, cnt + O_MW1);
    k_absmax<<<32, 256, 0, stream>>>(w2, WELEM, cnt + O_MW2);

    k_quant_w<<<(WELEM + 255) / 256, 256, 0, stream>>>(w1, cnt, O_MW1, wpk1, cnt + O_WB1);
    k_quant_w<<<(WELEM + 255) / 256, 256, 0, stream>>>(w2, cnt, O_MW2, wpk2, cnt + O_WB2);

    k_quant_x<<<(NOUT + 255) / 256, 256, 0, stream>>>(x, cnt, qx_pad);

    k_conv<0><<<PIX / 128, 256, 0, stream>>>(qx_pad, wpk1, cnt, cnt + O_MR1,
                                             g1, b1, m1, v1, (const int8_t*)nullptr, obuf);

    k_quant_q1<<<(NOUT + 255) / 256, 256, 0, stream>>>(obuf, cnt, q1_pad);

    k_conv<1><<<PIX / 128, 256, 0, stream>>>(q1_pad, wpk2, cnt, cnt + O_MOUT,
                                             g2, b2, m2, v2, qx_pad, tbuf);

    k_final<<<(NOUT + 255) / 256, 256, 0, stream>>>(tbuf, cnt, out);

    k_hm<<<1, 512, 0, stream>>>(cnt, out);
}
